// LocalFWLNet_34857954574876
// MI455X (gfx1250) — matrix operand order repaired and checked by reading, not compiled, not run on hardware
//
#include <hip/hip_runtime.h>
#include <hip/hip_bf16.h>

typedef __attribute__((ext_vector_type(16))) _Float16 v16h;
typedef __attribute__((ext_vector_type(8)))  _Float16 v8h;
typedef __attribute__((ext_vector_type(8)))  float    v8f;
typedef __attribute__((ext_vector_type(4)))  unsigned v4u;
typedef __attribute__((ext_vector_type(8)))  int      v8i;
typedef __attribute__((ext_vector_type(4)))  int      v4i;

constexpr int NODES = 768;
constexpr int HID   = 32;
constexpr int EDGES = 24576;
constexpr int NPAIR = 4096;
constexpr int NN2   = NODES * NODES;          // 589824
constexpr int WPR   = 24;                     // 768 bits / 32 words per bitset row
constexpr float EPSV = 1e-5f;

// padded LDS tile geometry (TDM pad: +16B after every 128B row)
constexpr int RSH   = 72;                     // LDS row stride in halves (64 + 8 pad)
constexpr int TILEB = 64 * RSH * 2;           // 9216 bytes per 64x64 tile
constexpr int BUFB  = 2 * TILEB;              // A + B per buffer = 18432 bytes

// ---------------- small prep kernels ----------------

__global__ void k_embed(const int* __restrict__ x, const float* __restrict__ emb,
                        float* __restrict__ h) {
  int t = blockIdx.x * blockDim.x + threadIdx.x;
  if (t >= NODES * HID) return;
  int i = t / HID, d = t % HID;
  h[t] = emb[x[i] * HID + d];
}

__global__ void k_edges(const int* __restrict__ ei, float* __restrict__ A,
                        unsigned* __restrict__ adjb) {
  int e = blockIdx.x * blockDim.x + threadIdx.x;
  if (e >= EDGES) return;
  int i = ei[e], j = ei[EDGES + e];
  A[i * NODES + j] = 1.0f;                              // .set semantics (dups ok)
  atomicOr(&adjb[i * WPR + (j >> 5)], 1u << (j & 31));
}

__global__ void k_diag(float* __restrict__ A) {
  int i = blockIdx.x * blockDim.x + threadIdx.x;
  if (i < NODES) A[i * NODES + i] += 1.0f;              // Ahat = A + I
}

__global__ void k_dinv(const float* __restrict__ A, float* __restrict__ dinv) {
  int i = blockIdx.x * blockDim.x + threadIdx.x;
  if (i >= NODES) return;
  float s = 0.f;
  for (int j = 0; j < NODES; ++j) s += A[i * NODES + j];
  dinv[i] = rsqrtf(s);
}

__global__ void k_scaleA(float* __restrict__ A, const float* __restrict__ dinv) {
  int t = blockIdx.x * blockDim.x + threadIdx.x;      // grid sized exactly NN2/256
  int i = t / NODES, j = t % NODES;
  A[t] *= dinv[i] * dinv[j];
}

// hg = h @ W   (N x 32) @ (32 x 32)
__global__ void k_hW(const float* __restrict__ h, const float* __restrict__ W,
                     float* __restrict__ hg) {
  int t = blockIdx.x * blockDim.x + threadIdx.x;
  int i = t / HID, d = t % HID;
  float s = 0.f;
  for (int c = 0; c < HID; ++c) s += h[i * HID + c] * W[c * HID + d];
  hg[t] = s;
}

// hout = An @ hg + b
__global__ void k_AnMM(const float* __restrict__ An, const float* __restrict__ hg,
                       const float* __restrict__ b, float* __restrict__ hout) {
  int t = blockIdx.x * blockDim.x + threadIdx.x;
  int i = t / HID, d = t % HID;
  float s = b[d];
  for (int k = 0; k < NODES; ++k) s += An[i * NODES + k] * hg[k * HID + d];
  hout[t] = s;
}

// graph_norm + relu, one block (256 thr) per channel d, in place
__global__ __launch_bounds__(256) void k_gnorm(float* __restrict__ h) {
  int d = blockIdx.x;
  __shared__ float red[256];
  __shared__ float meanS, rstdS;
  float s = 0.f;
  for (int i = threadIdx.x; i < NODES; i += 256) s += h[i * HID + d];
  red[threadIdx.x] = s; __syncthreads();
  for (int o = 128; o > 0; o >>= 1) {
    if (threadIdx.x < o) red[threadIdx.x] += red[threadIdx.x + o];
    __syncthreads();
  }
  if (threadIdx.x == 0) meanS = red[0] / (float)NODES;
  __syncthreads();
  float mean = meanS, ss = 0.f;
  for (int i = threadIdx.x; i < NODES; i += 256) {
    float v = h[i * HID + d] - mean; ss += v * v;
  }
  red[threadIdx.x] = ss; __syncthreads();
  for (int o = 128; o > 0; o >>= 1) {
    if (threadIdx.x < o) red[threadIdx.x] += red[threadIdx.x + o];
    __syncthreads();
  }
  if (threadIdx.x == 0) rstdS = rsqrtf(red[0] / (float)NODES + EPSV);
  __syncthreads();
  float rstd = rstdS;
  for (int i = threadIdx.x; i < NODES; i += 256) {
    float v = (h[i * HID + d] - mean) * rstd;
    h[i * HID + d] = v > 0.f ? v : 0.f;
  }
}

// per-edge MLPs, scatter-add into fp32 Xd/Md
__global__ void k_edge_mlp(const int* __restrict__ ei, const float* __restrict__ h,
                           const float* __restrict__ W1, const float* __restrict__ b1,
                           const float* __restrict__ W2, const float* __restrict__ b2,
                           float* __restrict__ Xd32, float* __restrict__ Md32) {
  int t = blockIdx.x * blockDim.x + threadIdx.x;
  int e = t / HID, d = t % HID;
  int i = ei[e], j = ei[EDGES + e];
  float s1 = b1[d], s2 = b2[d];
  for (int c = 0; c < HID; ++c) {
    float hv = h[i * HID + c];
    s1 += hv * W1[c * HID + d];
    s2 += hv * W2[c * HID + d];
  }
  for (int c = 0; c < HID; ++c) {
    float hv = h[j * HID + c];
    s1 += hv * W1[(HID + c) * HID + d];
    s2 += hv * W2[(HID + c) * HID + d];
  }
  s1 = fmaxf(s1, 0.f); s2 = fmaxf(s2, 0.f);
  atomicAdd(&Xd32[(size_t)(i * NODES + j) * HID + d], s1);
  atomicAdd(&Md32[(size_t)(i * NODES + j) * HID + d], s2);
}

// Xdh[d][i][k] = (f16) Xd32[i][k][d]
__global__ void k_convX(const float* __restrict__ Xd32, _Float16* __restrict__ Xdh) {
  int t = blockIdx.x * blockDim.x + threadIdx.x;   // t = i*N + k
  const float* src = Xd32 + (size_t)t * HID;
  for (int d = 0; d < HID; ++d)
    Xdh[(size_t)d * NN2 + t] = (_Float16)src[d];
}

// Mdh[d][j][k] = (f16) Md32[k][j][d]   (B stored transposed for fragment loads)
__global__ void k_convM(const float* __restrict__ Md32, _Float16* __restrict__ Mdh) {
  int t = blockIdx.x * blockDim.x + threadIdx.x;
  int j = t / NODES, k = t % NODES;
  const float* src = Md32 + ((size_t)k * NODES + j) * HID;
  for (int d = 0; d < HID; ++d)
    Mdh[(size_t)d * NN2 + (size_t)j * NODES + k] = (_Float16)src[d];
}

// maskbits = (A@A > 0) | A via 768-bit row bitsets
__global__ void k_mask(const unsigned* __restrict__ adjb, unsigned* __restrict__ mskb) {
  int i = blockIdx.x;
  int w = threadIdx.x;
  if (w >= WPR) return;
  unsigned m = adjb[i * WPR + w];
  for (int k = 0; k < NODES; ++k) {
    if (adjb[i * WPR + (k >> 5)] & (1u << (k & 31)))
      m |= adjb[k * WPR + w];
  }
  mskb[i * WPR + w] = m;
}

// ---------------- TDM helper: async 2-D f16 tile load, global -> LDS ----------------
// D# per ISA ch.8. pad_enable inserts 4 DWORDs (16B) of LDS padding after every
// 32 DWORDs (one 128B tile row) -> LDS row stride 144B, which spreads the 16
// fragment-load lanes across all 64 LDS banks (36r mod 64 distinct) instead of
// the 8-way conflicts a 128B stride produces.
__device__ __forceinline__ void tdm_load_2d(unsigned lds_byte_off, const void* gptr,
                                            unsigned tile_d0, unsigned tile_d1,
                                            unsigned tens_d0, unsigned tens_d1,
                                            unsigned stride0) {
  unsigned long long ga = (unsigned long long)(uintptr_t)gptr;
  v4u g0;
  g0[0] = 1u;                                             // count = 1 valid descriptor
  g0[1] = lds_byte_off;                                   // lds_addr (bytes)
  g0[2] = (unsigned)(ga & 0xFFFFFFFFu);                   // global_addr[31:0]
  g0[3] = (unsigned)((ga >> 32) & 0x01FFFFFFu) | (2u << 30); // addr[56:32] | type=2
  v8i g1;
  // data_size=1 (2B) | pad_enable | pad_interval=4 (32 DW) | pad_amount=3 (4 DW)
  g1[0] = (int)((1u << 16) | (1u << 20) | (4u << 22) | (3u << 25));
  g1[1] = (int)(tens_d0 << 16);                           // tensor_dim0[15:0]
  g1[2] = (int)((tens_d0 >> 16) | (tens_d1 << 16));       // dim0 hi | dim1 lo
  g1[3] = (int)((tens_d1 >> 16) | (tile_d0 << 16));       // dim1 hi | tile_dim0
  g1[4] = (int)tile_d1;                                   // tile_dim1, tile_dim2=0
  g1[5] = (int)stride0;                                   // tensor_dim0_stride[31:0]
  g1[6] = 0;
  g1[7] = 0;
  v4i gz = {0, 0, 0, 0};
#if __clang_major__ >= 23
  v8i gz8 = {0, 0, 0, 0, 0, 0, 0, 0};
  __builtin_amdgcn_tensor_load_to_lds(g0, g1, gz, gz, gz8, 0);
#else
  __builtin_amdgcn_tensor_load_to_lds(g0, g1, gz, gz, 0);
#endif
}

// ---------------- batched WMMA GEMM with TDM double-buffered LDS staging ----------
// C[d] (768x768 f32) = Xdh[d] ([i][k] f16) @ Mdh[d] stored transposed ([j][k] f16)
// Block = 4 waves, 64x64 output tile per d. Per 64-wide k-chunk the TDM stages
// A(64x64) + B(64x64) halves serving 32 WMMAs -> ~0.5KB/WMMA of global traffic.
__global__ __launch_bounds__(128) void k_gemm(const _Float16* __restrict__ Xdh,
                                              const _Float16* __restrict__ Mdh,
                                              float* __restrict__ C) {
  extern __shared__ _Float16 sm[];   // 2 buffers x (A + B padded tiles) = 36864 B
  int d  = blockIdx.z;
  int i0 = blockIdx.x * 64;
  int j0 = blockIdx.y * 64;
  int lane = threadIdx.x & 31;
  int wv   = threadIdx.x >> 5;       // wave id 0..3 -> i sub-tile
  int lm = lane & 15;
  int hi = lane >> 4;

  const _Float16* Ad = Xdh + (size_t)d * NN2 + (size_t)i0 * NODES;
  const _Float16* Bd = Mdh + (size_t)d * NN2 + (size_t)j0 * NODES;
  unsigned ldsbase = (unsigned)(uintptr_t)(&sm[0]);   // LDS byte offset of sm

  v8f acc0 = {}, acc1 = {}, acc2 = {}, acc3 = {};

  // prologue: stage chunk 0 into buffer 0 (issued once by wave 0; EXEC ignored)
  if (threadIdx.x == 0) {
    tdm_load_2d(ldsbase,         Ad, 64, 64, NODES, NODES, NODES);
    tdm_load_2d(ldsbase + TILEB, Bd, 64, 64, NODES, NODES, NODES);
  }

  int cur = 0;
  for (int step = 0; step < NODES / 64; ++step) {
    if (threadIdx.x == 0) __builtin_amdgcn_s_wait_tensorcnt(0);
    __syncthreads();   // buffer 'cur' ready for all; all done reading 'cur^1'
    if (threadIdx.x == 0 && step + 1 < NODES / 64) {
      unsigned nb = ldsbase + (unsigned)(cur ^ 1) * (unsigned)BUFB;
      tdm_load_2d(nb,         Ad + (step + 1) * 64, 64, 64, NODES, NODES, NODES);
      tdm_load_2d(nb + TILEB, Bd + (step + 1) * 64, 64, 64, NODES, NODES, NODES);
    }
    const _Float16* sA = sm + cur * (BUFB / 2);                // [64][RSH] halves
    const _Float16* sB = sm + cur * (BUFB / 2) + (TILEB / 2);  // [64][RSH] halves
#pragma unroll
    for (int kk = 0; kk < 64; kk += 32) {
      const _Float16* ap = sA + (wv * 16 + lm) * RSH + kk + hi * 16;
      v8h alo = *(const v8h*)ap;
      v8h ahi = *(const v8h*)(ap + 8);
      v16h a;
#pragma unroll
      for (int q = 0; q < 8; ++q) { a[q] = alo[q]; a[q + 8] = ahi[q]; }
      v16h b0, b1, b2, b3;
      {
        const _Float16* bp0 = sB + ( 0 + lm) * RSH + kk + hi * 16;
        const _Float16* bp1 = sB + (16 + lm) * RSH + kk + hi * 16;
        const _Float16* bp2 = sB + (32 + lm) * RSH + kk + hi * 16;
        const _Float16* bp3 = sB + (48 + lm) * RSH + kk + hi * 16;
        v8h t0 = *(const v8h*)bp0, t1 = *(const v8h*)(bp0 + 8);
        v8h u0 = *(const v8h*)bp1, u1 = *(const v8h*)(bp1 + 8);
        v8h w0 = *(const v8h*)bp2, w1 = *(const v8h*)(bp2 + 8);
        v8h y0 = *(const v8h*)bp3, y1 = *(const v8h*)(bp3 + 8);
#pragma unroll
        for (int q = 0; q < 8; ++q) {
          b0[q] = t0[q]; b0[q + 8] = t1[q];
          b1[q] = u0[q]; b1[q + 8] = u1[q];
          b2[q] = w0[q]; b2[q + 8] = w1[q];
          b3[q] = y0[q]; b3[q + 8] = y1[q];
        }
      }
      acc0 = __builtin_amdgcn_wmma_f32_16x16x32_f16(false, a, false, b0, (short)0, acc0, false, false);
      acc1 = __builtin_amdgcn_wmma_f32_16x16x32_f16(false, a, false, b1, (short)0, acc1, false, false);
      acc2 = __builtin_amdgcn_wmma_f32_16x16x32_f16(false, a, false, b2, (short)0, acc2, false, false);
      acc3 = __builtin_amdgcn_wmma_f32_16x16x32_f16(false, a, false, b3, (short)0, acc3, false, false);
    }
    cur ^= 1;
  }

  // C/D map: lane = (n, hi half of M), VGPR r -> M = hi*8 + r
  float* Cd = C + (size_t)d * NN2;
  int ib = i0 + wv * 16 + hi * 8;
#pragma unroll
  for (int r = 0; r < 8; ++r) {
    size_t row = (size_t)(ib + r) * NODES;
    Cd[row + j0 +  0 + lm] = acc0[r];
    Cd[row + j0 + 16 + lm] = acc1[r];
    Cd[row + j0 + 32 + lm] = acc2[r];
    Cd[row + j0 + 48 + lm] = acc3[r];
  }
}

// ---------------- z = [C, ind] @ W3 + b3, masked stats, norm ----------------
// red[0..31]=sum(z*m), red[32]=cnt, red[33..64]=var sums

__global__ __launch_bounds__(256) void k_z1(float* __restrict__ C,
                                            const unsigned* __restrict__ adjb,
                                            const unsigned* __restrict__ mskb,
                                            const float* __restrict__ W3,
                                            const float* __restrict__ b3,
                                            float* __restrict__ red) {
  __shared__ float sW[33 * 32];
  __shared__ float sb[32];
  __shared__ float ssum[33];
  for (int t = threadIdx.x; t < 33 * 32; t += 256) sW[t] = W3[t];
  if (threadIdx.x < 32) sb[threadIdx.x] = b3[threadIdx.x];
  if (threadIdx.x < 33) ssum[threadIdx.x] = 0.f;
  __syncthreads();
  int t = blockIdx.x * 256 + threadIdx.x;          // NN2 divisible by 256
  int i = t / NODES, j = t % NODES;
  float c[32];
#pragma unroll
  for (int d = 0; d < 32; ++d) c[d] = C[(size_t)d * NN2 + t];
  float af = ((adjb[i * WPR + (j >> 5)] >> (j & 31)) & 1u) ? 1.f : 0.f;
  bool  mk = ((mskb[i * WPR + (j >> 5)] >> (j & 31)) & 1u) != 0u;
  float z[32];
#pragma unroll
  for (int d2 = 0; d2 < 32; ++d2) {
    float s = sb[d2] + af * sW[32 * 32 + d2];
    for (int d = 0; d < 32; ++d) s += c[d] * sW[d * 32 + d2];
    z[d2] = s;
  }
#pragma unroll
  for (int d = 0; d < 32; ++d) C[(size_t)d * NN2 + t] = z[d];
  if (mk) {
    for (int d = 0; d < 32; ++d) atomicAdd(&ssum[d], z[d]);
    atomicAdd(&ssum[32], 1.0f);
  }
  __syncthreads();
  if (threadIdx.x < 33) atomicAdd(&red[threadIdx.x], ssum[threadIdx.x]);
}

__global__ __launch_bounds__(256) void k_z2(const float* __restrict__ C,
                                            const unsigned* __restrict__ mskb,
                                            const float* __restrict__ red,
                                            float* __restrict__ redv) {
  __shared__ float ssum[32];
  if (threadIdx.x < 32) ssum[threadIdx.x] = 0.f;
  __syncthreads();
  int t = blockIdx.x * 256 + threadIdx.x;
  int i = t / NODES, j = t % NODES;
  bool mk = ((mskb[i * WPR + (j >> 5)] >> (j & 31)) & 1u) != 0u;
  float inv = 1.f / red[32];
  if (mk) {
    for (int d = 0; d < 32; ++d) {
      float v = C[(size_t)d * NN2 + t] - red[d] * inv;
      atomicAdd(&ssum[d], v * v);
    }
  }
  __syncthreads();
  if (threadIdx.x < 32) atomicAdd(&redv[threadIdx.x], ssum[threadIdx.x]);
}

__global__ __launch_bounds__(256) void k_z3(float* __restrict__ C,
                                            const float* __restrict__ red) {
  int t = blockIdx.x * 256 + threadIdx.x;
  float inv = 1.f / red[32];
#pragma unroll
  for (int d = 0; d < 32; ++d) {
    float mean = red[d] * inv;
    float var  = red[33 + d] * inv;
    float v = (C[(size_t)d * NN2 + t] - mean) * rsqrtf(var + EPSV);
    C[(size_t)d * NN2 + t] = v > 0.f ? v : 0.f;
  }
}

__global__ void k_out(const float* __restrict__ z, const float* __restrict__ hfin,
                      const int* __restrict__ pos, const unsigned* __restrict__ mskb,
                      const float* __restrict__ linW, const float* __restrict__ linb,
                      float* __restrict__ out) {
  int p = blockIdx.x * blockDim.x + threadIdx.x;
  if (p >= NPAIR) return;
  int i = pos[2 * p], j = pos[2 * p + 1];
  float m = ((mskb[i * WPR + (j >> 5)] >> (j & 31)) & 1u) ? 1.f : 0.f;
  float acc = linb[0];
  for (int d = 0; d < 32; ++d) {
    float s = z[(size_t)d * NN2 + (size_t)i * NODES + j] *
              z[(size_t)d * NN2 + (size_t)j * NODES + i] * m;
    acc += s * linW[d];
    acc += hfin[i * HID + d] * hfin[j * HID + d] * linW[32 + d];
  }
  out[p] = acc;
}

// ---------------- launcher ----------------

extern "C" void kernel_launch(void* const* d_in, const int* in_sizes, int n_in,
                              void* d_out, int out_size, void* d_ws, size_t ws_size,
                              hipStream_t stream) {
  const int*   x    = (const int*)  d_in[0];
  const int*   ei   = (const int*)  d_in[1];
  const int*   pos  = (const int*)  d_in[2];
  const float* emb  = (const float*)d_in[3];
  const float* gcnW = (const float*)d_in[4];
  const float* gcnB = (const float*)d_in[5];
  const float* W1   = (const float*)d_in[6];
  const float* b1   = (const float*)d_in[7];
  const float* W2   = (const float*)d_in[8];
  const float* b2   = (const float*)d_in[9];
  const float* W3   = (const float*)d_in[10];
  const float* b3   = (const float*)d_in[11];
  const float* linW = (const float*)d_in[12];
  const float* linb = (const float*)d_in[13];
  float* out = (float*)d_out;

  char* ws = (char*)d_ws;
  size_t off = 0;
  auto alloc = [&](size_t bytes) -> char* {
    char* p = ws + off;
    off = (off + bytes + 255) & ~(size_t)255;
    return p;
  };
  float*    An   = (float*)   alloc((size_t)NN2 * 4);
  unsigned* adjb = (unsigned*)alloc((size_t)NODES * WPR * 4);
  unsigned* mskb = (unsigned*)alloc((size_t)NODES * WPR * 4);
  float*    h0   = (float*)   alloc((size_t)NODES * HID * 4);
  float*    h1   = (float*)   alloc((size_t)NODES * HID * 4);
  float*    hg   = (float*)   alloc((size_t)NODES * HID * 4);
  float*    dinv = (float*)   alloc((size_t)NODES * 4);
  float*    red  = (float*)   alloc(256);
  float*    Xd32 = (float*)   alloc((size_t)NN2 * HID * 4);
  float*    Md32 = (float*)   alloc((size_t)NN2 * HID * 4);
  _Float16* Xdh  = (_Float16*)alloc((size_t)NN2 * HID * 2);
  _Float16* Mdh  = (_Float16*)alloc((size_t)NN2 * HID * 2);
  float*    C    = Xd32;   // reuse: Xd32 dead after k_convX

  hipMemsetAsync(An,   0, (size_t)NN2 * 4, stream);
  hipMemsetAsync(adjb, 0, (size_t)NODES * WPR * 4, stream);
  hipMemsetAsync(red,  0, 256, stream);
  hipMemsetAsync(Xd32, 0, (size_t)NN2 * HID * 4, stream);
  hipMemsetAsync(Md32, 0, (size_t)NN2 * HID * 4, stream);

  k_embed <<<(NODES * HID + 255) / 256, 256, 0, stream>>>(x, emb, h0);
  k_edges <<<(EDGES + 255) / 256,      256, 0, stream>>>(ei, An, adjb);
  k_diag  <<<(NODES + 255) / 256,      256, 0, stream>>>(An);
  k_dinv  <<<(NODES + 255) / 256,      256, 0, stream>>>(An, dinv);
  k_scaleA<<<NN2 / 256,                256, 0, stream>>>(An, dinv);

  float* hc = h0; float* hn = h1;
  for (int l = 0; l < 2; ++l) {
    k_hW   <<<(NODES * HID) / 256, 256, 0, stream>>>(hc, gcnW + l * HID * HID, hg);
    k_AnMM <<<(NODES * HID) / 256, 256, 0, stream>>>(An, hg, gcnB + l * HID, hn);
    k_gnorm<<<HID, 256, 0, stream>>>(hn);
    float* tmp = hc; hc = hn; hn = tmp;
  }

  k_edge_mlp<<<(EDGES * HID) / 256, 256, 0, stream>>>(ei, hc, W1, b1, W2, b2, Xd32, Md32);
  k_convX<<<NN2 / 256, 256, 0, stream>>>(Xd32, Xdh);
  k_convM<<<NN2 / 256, 256, 0, stream>>>(Md32, Mdh);
  k_mask <<<NODES, 32, 0, stream>>>(adjb, mskb);

  dim3 gemmGrid(NODES / 64, NODES / 64, HID);   // 12 x 12 x 32 blocks, 4 waves each
  k_gemm<<<gemmGrid, 128, 2 * BUFB, stream>>>(Xdh, Mdh, C);

  k_z1<<<NN2 / 256, 256, 0, stream>>>(C, adjb, mskb, W3, b3, red);
  k_z2<<<NN2 / 256, 256, 0, stream>>>(C, mskb, red, red + 33);
  k_z3<<<NN2 / 256, 256, 0, stream>>>(C, red);
  k_out<<<(NPAIR + 255) / 256, 256, 0, stream>>>(C, hc, pos, mskb, linW, linb, out);
}
